// Net_2937757630586
// MI455X (gfx1250) — matrix operand order repaired and checked by reading, not compiled, not run on hardware
//
#include <hip/hip_runtime.h>
#include <math.h>

typedef __attribute__((ext_vector_type(16))) _Float16 v16h;
typedef __attribute__((ext_vector_type(8)))  float    v8f;

#define THREADS 256

// ---------- init: deg = 1 (self-loop), accumulators = 0 ----------
__global__ __launch_bounds__(THREADS) void k_init(float* __restrict__ deg,
                                                  float* __restrict__ A1,
                                                  float* __restrict__ A2, int n) {
  int idx = blockIdx.x * THREADS + threadIdx.x;
  if (idx < n * 16) A1[idx] = 0.0f;
  if (idx < n * 7)  A2[idx] = 0.0f;
  if (idx < n)      deg[idx] = 1.0f;
}

// ---------- degree count over dst ----------
__global__ __launch_bounds__(THREADS) void k_deg(const int* __restrict__ dst,
                                                 float* __restrict__ deg, int E) {
  int e = blockIdx.x * THREADS + threadIdx.x;
  if (e < E) atomicAdd(&deg[dst[e]], 1.0f);
}

// ---------- dinv = rsqrt(deg), in place ----------
__global__ __launch_bounds__(THREADS) void k_dinv(float* __restrict__ deg, int n) {
  int i = blockIdx.x * THREADS + threadIdx.x;
  if (i < n) deg[i] = rsqrtf(deg[i]);   // deg >= 1 always
}

// ---------- GEMM1: H1[n x 16] = x[n x 512] @ W1[512 x 16] via v_wmma_f32_16x16x32_f16 ----------
// B fragments for all 16 K-steps staged once per block into LDS as f16 in the
// exact per-lane WMMA layout: bsh[kstep][lane][t], 16*32*16 halves = 16 KB.
__global__ __launch_bounds__(THREADS) void k_gemm1(const float* __restrict__ x,
                                                   const float* __restrict__ W1,
                                                   float* __restrict__ H1, int n) {
  __shared__ __align__(32) _Float16 bsh[16 * 32 * 16];

  // cooperative fill: b[t] for (kstep, lane) is W1[(kstep*32 + (t>>3)*16 + (lane>>4)*8 + (t&7))*16 + (lane&15)]
  for (int idx = threadIdx.x; idx < 16 * 32 * 16; idx += THREADS) {
    int kstep = idx >> 9;          // /512
    int ln    = (idx >> 4) & 31;
    int t     = idx & 15;
    int kk    = kstep * 32 + ((t >> 3) << 4) + ((ln >> 4) << 3) + (t & 7);
    int col   = ln & 15;
    bsh[idx] = (_Float16)W1[kk * 16 + col];
  }
  __syncthreads();

  const int lane = threadIdx.x & 31;
  const int wave = threadIdx.x >> 5;
  const int ntiles = (n + 15) >> 4;
  const int tile = blockIdx.x * 8 + wave;
  if (tile >= ntiles) return;                 // wave-uniform: EXEC all-1 inside

  const int row0 = tile << 4;
  const int mlane = lane & 15;                // row (A) / col (B,C,D) within tile
  const int hb = lane >> 4;                   // K-half selector
  int row = row0 + mlane;
  int rowc = row < n ? row : n - 1;           // clamp loads
  const float* xrow = x + (size_t)rowc * 512;
  const v16h* bfrag = (const v16h*)&bsh[lane * 16];   // + kstep*512 halves

  v8f acc = {};
  for (int k0 = 0; k0 < 512; k0 += 32) {
    // A fragment: 16x32 f16 (two 32B vector loads of x per lane, cvt to f16)
    v16h a;
#pragma unroll
    for (int k = 0; k < 8; ++k) {
      a[k]     = (_Float16)xrow[k0 + hb * 8 + k];
      a[8 + k] = (_Float16)xrow[k0 + 16 + hb * 8 + k];
    }
    v16h b = bfrag[(k0 >> 5) * 32];   // v16h stride 32 elems*? no: index in v16h units
    acc = __builtin_amdgcn_wmma_f32_16x16x32_f16(false, a, false, b,
                                                 (short)0, acc, false, false);
  }

  // D store: VGPR r -> row (row0 + r + 8*hb), col mlane
  if (row0 + 16 <= n) {             // uniform fast path: no per-store predication
#pragma unroll
    for (int r = 0; r < 8; ++r)
      H1[(size_t)(row0 + r + hb * 8) * 16 + mlane] = acc[r];
  } else {
#pragma unroll
    for (int r = 0; r < 8; ++r) {
      int m = row0 + r + hb * 8;
      if (m < n) H1[(size_t)m * 16 + mlane] = acc[r];
    }
  }
}

// ---------- edge aggregation: A[dst] += dinv[s]*dinv[d] * H[src], F features ----------
template <int F>
__global__ __launch_bounds__(THREADS) void k_agg(const int* __restrict__ src,
                                                 const int* __restrict__ dst,
                                                 const float* __restrict__ dinv,
                                                 const float* __restrict__ H,
                                                 float* __restrict__ A, int E) {
  long long idx = (long long)blockIdx.x * THREADS + threadIdx.x;
  if (idx >= (long long)E * F) return;
  int e = (int)(idx / F);
  int j = (int)(idx - (long long)e * F);
  int s = src[e], d = dst[e];
  float w = dinv[s] * dinv[d];
  atomicAdd(&A[(long long)d * F + j], w * H[(long long)s * F + j]);
}

// ---------- self-loop + bias + ReLU (layer 1), Z written in place over A1 ----------
__global__ __launch_bounds__(THREADS) void k_relu(const float* __restrict__ H1,
                                                  const float* __restrict__ dinv,
                                                  const float* __restrict__ b1,
                                                  float* __restrict__ A1, int n) {
  int idx = blockIdx.x * THREADS + threadIdx.x;
  if (idx >= n * 16) return;
  int i = idx >> 4, j = idx & 15;
  float di = dinv[i];
  float v = A1[idx] + di * di * H1[idx] + b1[j];
  A1[idx] = v > 0.0f ? v : 0.0f;
}

// ---------- GEMM2: H2[n x 7] = Z[n x 16] @ W2[16 x 7], K padded to 32, N padded to 16 ----------
__global__ __launch_bounds__(THREADS) void k_gemm2(const float* __restrict__ Z,
                                                   const float* __restrict__ W2,
                                                   float* __restrict__ H2, int n) {
  const int lane = threadIdx.x & 31;
  const int wave = threadIdx.x >> 5;
  const int ntiles = (n + 15) >> 4;
  const int tile = blockIdx.x * 8 + wave;
  if (tile >= ntiles) return;

  const int row0 = tile << 4;
  const int mlane = lane & 15;
  const int hb = lane >> 4;
  int row = row0 + mlane;
  int rowc = row < n ? row : n - 1;
  const float* zrow = Z + (size_t)rowc * 16;

  v16h a;
#pragma unroll
  for (int k = 0; k < 8; ++k) {
    a[k]     = (_Float16)zrow[hb * 8 + k];   // K = hb*8+k in [0,16)
    a[8 + k] = (_Float16)0.0f;               // K >= 16 : zero pad
  }
  v16h b;
#pragma unroll
  for (int t = 0; t < 16; ++t) {
    int kk = ((t >> 3) << 4) + hb * 8 + (t & 7);
    b[t] = (kk < 16 && mlane < 7) ? (_Float16)W2[kk * 7 + mlane] : (_Float16)0.0f;
  }
  v8f acc = {};
  acc = __builtin_amdgcn_wmma_f32_16x16x32_f16(false, a, false, b,
                                               (short)0, acc, false, false);

  if (row0 + 16 <= n) {
#pragma unroll
    for (int r = 0; r < 8; ++r)
      if (mlane < 7) H2[(size_t)(row0 + r + hb * 8) * 7 + mlane] = acc[r];
  } else {
#pragma unroll
    for (int r = 0; r < 8; ++r) {
      int m = row0 + r + hb * 8;
      if (m < n && mlane < 7) H2[(size_t)m * 7 + mlane] = acc[r];
    }
  }
}

// ---------- self-loop + bias + log_softmax (layer 2) ----------
__global__ __launch_bounds__(THREADS) void k_final(const float* __restrict__ A2,
                                                   const float* __restrict__ H2,
                                                   const float* __restrict__ dinv,
                                                   const float* __restrict__ b2,
                                                   float* __restrict__ out, int n) {
  int i = blockIdx.x * THREADS + threadIdx.x;
  if (i >= n) return;
  float w = dinv[i] * dinv[i];
  float v[7];
  float m = -INFINITY;
#pragma unroll
  for (int j = 0; j < 7; ++j) {
    v[j] = A2[(size_t)i * 7 + j] + w * H2[(size_t)i * 7 + j] + b2[j];
    m = fmaxf(m, v[j]);
  }
  float s = 0.0f;
#pragma unroll
  for (int j = 0; j < 7; ++j) s += expf(v[j] - m);
  float l = logf(s);
#pragma unroll
  for (int j = 0; j < 7; ++j) out[(size_t)i * 7 + j] = v[j] - m - l;
}

extern "C" void kernel_launch(void* const* d_in, const int* in_sizes, int n_in,
                              void* d_out, int out_size, void* d_ws, size_t ws_size,
                              hipStream_t stream) {
  const float* x  = (const float*)d_in[0];
  const int*   ei = (const int*)d_in[1];
  const float* W1 = (const float*)d_in[2];
  const float* b1 = (const float*)d_in[3];
  const float* W2 = (const float*)d_in[4];
  const float* b2 = (const float*)d_in[5];
  float* out = (float*)d_out;

  const int n = in_sizes[0] / 512;
  const int E = in_sizes[1] / 2;
  const int* src = ei;
  const int* dst = ei + E;

  // workspace layout (floats): deg/dinv | H1 | A1(=Z) | H2 | A2
  float* deg = (float*)d_ws;
  float* H1  = deg + n;
  float* A1  = H1 + (size_t)n * 16;
  float* H2  = A1 + (size_t)n * 16;
  float* A2  = H2 + (size_t)n * 7;

  const int ntiles = (n + 15) / 16;
  const int gemm_blocks = (ntiles + 7) / 8;

  k_init <<<(n * 16 + THREADS - 1) / THREADS, THREADS, 0, stream>>>(deg, A1, A2, n);
  k_deg  <<<(E + THREADS - 1) / THREADS, THREADS, 0, stream>>>(dst, deg, E);
  k_dinv <<<(n + THREADS - 1) / THREADS, THREADS, 0, stream>>>(deg, n);
  k_gemm1<<<gemm_blocks, THREADS, 0, stream>>>(x, W1, H1, n);
  {
    long long tot = (long long)E * 16;
    k_agg<16><<<(int)((tot + THREADS - 1) / THREADS), THREADS, 0, stream>>>(src, dst, deg, H1, A1, E);
  }
  k_relu <<<(n * 16 + THREADS - 1) / THREADS, THREADS, 0, stream>>>(H1, deg, b1, A1, n);
  k_gemm2<<<gemm_blocks, THREADS, 0, stream>>>(A1, W2, H2, n);
  {
    long long tot = (long long)E * 7;
    k_agg<7><<<(int)((tot + THREADS - 1) / THREADS), THREADS, 0, stream>>>(src, dst, deg, H2, A2, E);
  }
  k_final<<<(n + THREADS - 1) / THREADS, THREADS, 0, stream>>>(A2, H2, deg, b2, out, n);
}
